// MoeLayer_80882824118313
// MI455X (gfx1250) — matrix operand order repaired and checked by reading, not compiled, not run on hardware
//
#include <hip/hip_runtime.h>

#define NTOK 8192
#define DDIM 1024
#define HDIM 4096
#define NEXP 8

typedef __attribute__((ext_vector_type(16))) __bf16 v16bf;
typedef __attribute__((ext_vector_type(8)))  float  v8f;

union FragU {
  v16bf v;
  uint4 u[2];
};

// round-half-up f32->bf16, two at a time: 2x v_add + 1x v_perm
__device__ __forceinline__ unsigned int pkbf(float a, float b) {
  unsigned int ua = __float_as_uint(a) + 0x8000u;
  unsigned int ub = __float_as_uint(b) + 0x8000u;
  // result = [hi16(ub) : hi16(ua)] ; sel bytes: 2,3 from ua (S1), 6,7 from ub (S0)
  return __builtin_amdgcn_perm(ub, ua, 0x07060302u);
}
__device__ __forceinline__ unsigned short f32_to_bf16(float f) {
  return (unsigned short)((__float_as_uint(f) + 0x8000u) >> 16);
}
__device__ __forceinline__ float gelu_tanh(float v) {
  float t = tanhf(0.7978845608028654f * (v + 0.044715f * v * v * v));
  return 0.5f * v * (1.0f + t);
}

// ---------------------------------------------------------------- zero
__global__ __launch_bounds__(256) void zero_out_kernel(float* __restrict__ out,
                                                       int* __restrict__ counts,
                                                       int total) {
  for (int i = blockIdx.x * 256 + threadIdx.x; i < total; i += gridDim.x * 256)
    out[i] = 0.0f;
  if (blockIdx.x == 0 && threadIdx.x < NEXP) counts[threadIdx.x] = 0;
}

// ---------------------------------------------------------------- gating
__global__ __launch_bounds__(256) void gate_topk_kernel(
    const float* __restrict__ x, const float* __restrict__ gw,
    const float* __restrict__ gb, int* __restrict__ counts,
    int* __restrict__ tok_list, float* __restrict__ wgt_list) {
  const int lane = threadIdx.x & 31;
  const int wid  = threadIdx.x >> 5;
  const int n    = blockIdx.x * 8 + wid;
  if (n >= NTOK) return;

  float acc[NEXP];
#pragma unroll
  for (int e = 0; e < NEXP; ++e) acc[e] = 0.0f;

  for (int d = lane; d < DDIM; d += 32) {
    float xv = x[(size_t)n * DDIM + d];
#pragma unroll
    for (int e = 0; e < NEXP; ++e) acc[e] += xv * gw[d * NEXP + e];
  }
#pragma unroll
  for (int off = 16; off > 0; off >>= 1) {
#pragma unroll
    for (int e = 0; e < NEXP; ++e) acc[e] += __shfl_down(acc[e], off, 32);
  }
  if (lane == 0) {
#pragma unroll
    for (int e = 0; e < NEXP; ++e) acc[e] += gb[e];
    int i0 = 0;
#pragma unroll
    for (int e = 1; e < NEXP; ++e)
      if (acc[e] > acc[i0]) i0 = e;
    int i1 = (i0 == 0) ? 1 : 0;
#pragma unroll
    for (int e = 0; e < NEXP; ++e)
      if (e != i0 && acc[e] > acc[i1]) i1 = e;
    float p1 = 1.0f / (1.0f + __expf(acc[i0] - acc[i1]));
    float p0 = 1.0f - p1;
    int s0 = atomicAdd(&counts[i0], 1);
    tok_list[i0 * NTOK + s0] = n;
    wgt_list[i0 * NTOK + s0] = p0;
    int s1 = atomicAdd(&counts[i1], 1);
    tok_list[i1 * NTOK + s1] = n;
    wgt_list[i1 * NTOK + s1] = p1;
  }
}

// ---------------------------------------------------------------- prefix
__global__ void prefix_kernel(const int* __restrict__ counts,
                              int* __restrict__ offsets) {
  if (threadIdx.x == 0 && blockIdx.x == 0) {
    int s = 0;
    for (int e = 0; e < NEXP; ++e) { offsets[e] = s; s += counts[e]; }
  }
}

// ---------------------------------------------------------------- GEMM1 + GELU
// h[slot, nbase..+255] = gelu( x[tok] @ w1[e] + b1[e] ), bf16 out.
// Software-pipelined: global loads for tile k+1 in flight during compute of k.
__global__ __launch_bounds__(256) void gemm1_gelu_kernel(
    const float* __restrict__ x, const float* __restrict__ w1,
    const float* __restrict__ b1, const int* __restrict__ counts,
    const int* __restrict__ offsets, const int* __restrict__ tok_list,
    unsigned short* __restrict__ hbuf) {
  __shared__ unsigned short As[128 * 40];  // [m][k], stride 40 halves
  __shared__ unsigned short Bs[256 * 40];  // [n][k] (w1 tile transposed)
  __shared__ int tokS[128];

  const int e     = blockIdx.z;
  const int mt    = blockIdx.y;
  const int nbase = blockIdx.x * 256;
  const int rows  = counts[e];
  if (mt * 128 >= rows) return;
  const int obase = offsets[e];

  const int tid   = threadIdx.x;
  const int lane  = tid & 31;
  const int wave  = tid >> 5;
  const int waveM = (wave >> 2) * 64;
  const int waveN = (wave & 3) * 64;

  if (tid < 128) {
    int gr = mt * 128 + tid;
    tokS[tid] = (gr < rows) ? tok_list[e * NTOK + gr] : -1;
  }
  __syncthreads();

  v8f acc[4][4];
  const v8f vzero = {0.f, 0.f, 0.f, 0.f, 0.f, 0.f, 0.f, 0.f};
#pragma unroll
  for (int i = 0; i < 4; ++i)
#pragma unroll
    for (int j = 0; j < 4; ++j) acc[i][j] = vzero;

  const int aj = tid & 7;         // 4-float chunk within a row
  const int ar = tid >> 3;        // row base (0..31)
  const int bq = tid >> 6;        // k-quad base (0..3)
  const int bc = (tid & 63) * 4;  // col base

  float4 aReg[4];
  float4 bReg[2][4];

  auto loadA = [&](int k0) {
#pragma unroll
    for (int rr = 0; rr < 4; ++rr) {
      int r = ar + rr * 32;
      int tok = tokS[r];
      float4 v; v.x = 0.f; v.y = 0.f; v.z = 0.f; v.w = 0.f;
      if (tok >= 0) v = *(const float4*)&x[(size_t)tok * DDIM + k0 + 4 * aj];
      aReg[rr] = v;
    }
  };
  auto loadB = [&](int k0) {
#pragma unroll
    for (int it = 0; it < 2; ++it) {
      int kq = bq + it * 4;
#pragma unroll
      for (int kk = 0; kk < 4; ++kk)
        bReg[it][kk] =
            *(const float4*)&w1[((size_t)e * DDIM + (k0 + kq * 4 + kk)) * HDIM + nbase + bc];
    }
  };
  auto storeA = [&]() {
#pragma unroll
    for (int rr = 0; rr < 4; ++rr) {
      int r = ar + rr * 32;
      uint2 p;
      p.x = pkbf(aReg[rr].x, aReg[rr].y);
      p.y = pkbf(aReg[rr].z, aReg[rr].w);
      *(uint2*)&As[r * 40 + 4 * aj] = p;
    }
  };
  auto storeB = [&]() {
#pragma unroll
    for (int it = 0; it < 2; ++it) {
      int kq = bq + it * 4;
      const float* f0 = (const float*)&bReg[it][0];
      const float* f1 = (const float*)&bReg[it][1];
      const float* f2 = (const float*)&bReg[it][2];
      const float* f3 = (const float*)&bReg[it][3];
#pragma unroll
      for (int j = 0; j < 4; ++j) {
        uint2 p;
        p.x = pkbf(f0[j], f1[j]);  // k, k+1
        p.y = pkbf(f2[j], f3[j]);  // k+2, k+3
        *(uint2*)&Bs[(bc + j) * 40 + kq * 4] = p;
      }
    }
  };

  loadA(0);
  loadB(0);
  for (int k0 = 0; k0 < DDIM; k0 += 32) {
    __syncthreads();
    storeA();
    storeB();
    __syncthreads();
    if (k0 + 32 < DDIM) { loadA(k0 + 32); loadB(k0 + 32); }  // in flight during WMMA

    FragU af[4], bfr[4];
    const int ml  = lane & 15;
    const int kba = (lane < 16) ? 0 : 8;
    const int kbb = (lane < 16) ? 0 : 8;
#pragma unroll
    for (int t = 0; t < 4; ++t) {
      int m = waveM + t * 16 + ml;
      af[t].u[0] = *(const uint4*)&As[m * 40 + kba];
      af[t].u[1] = *(const uint4*)&As[m * 40 + kba + 16];
      int n = waveN + t * 16 + ml;
      bfr[t].u[0] = *(const uint4*)&Bs[n * 40 + kbb];
      bfr[t].u[1] = *(const uint4*)&Bs[n * 40 + kbb + 16];
    }
#pragma unroll
    for (int ti = 0; ti < 4; ++ti)
#pragma unroll
      for (int tj = 0; tj < 4; ++tj)
        acc[ti][tj] = __builtin_amdgcn_wmma_f32_16x16x32_bf16(
            false, af[ti].v, false, bfr[tj].v, (short)0, acc[ti][tj], false, false);
  }

  const int ncol = lane & 15;
  const int mofs = (lane < 16) ? 0 : 8;
#pragma unroll
  for (int tj = 0; tj < 4; ++tj) {
    int ng     = nbase + waveN + tj * 16 + ncol;
    float bias = b1[(size_t)e * HDIM + ng];
#pragma unroll
    for (int ti = 0; ti < 4; ++ti) {
#pragma unroll
      for (int r = 0; r < 8; ++r) {
        int mloc = waveM + ti * 16 + r + mofs;
        int mg   = mt * 128 + mloc;
        if (mg < rows) {
          float v = gelu_tanh(acc[ti][tj][r] + bias);
          hbuf[(size_t)(obase + mg) * HDIM + ng] = f32_to_bf16(v);
        }
      }
    }
  }
}

// ---------------------------------------------------------------- GEMM2 + scatter
// out[tok] += w * (h[slot] @ w2[e] + b2[e])
// A operand (bf16 h rows) copied with CDNA5 async global->LDS, double-buffered.
__global__ __launch_bounds__(256) void gemm2_scatter_kernel(
    const unsigned short* __restrict__ hbuf, const float* __restrict__ w2,
    const float* __restrict__ b2, const int* __restrict__ counts,
    const int* __restrict__ offsets, const int* __restrict__ tok_list,
    const float* __restrict__ wgt_list, float* __restrict__ out) {
  __shared__ unsigned short As2[2][128 * 40];
  __shared__ unsigned short Bs[256 * 40];
  __shared__ int   tokS[128];
  __shared__ float wgtS[128];

  const int e     = blockIdx.z;
  const int mt    = blockIdx.y;
  const int nbase = blockIdx.x * 256;
  const int rows  = counts[e];
  if (mt * 128 >= rows) return;
  const int obase = offsets[e];

  const int tid   = threadIdx.x;
  const int lane  = tid & 31;
  const int wave  = tid >> 5;
  const int waveM = (wave >> 2) * 64;
  const int waveN = (wave & 3) * 64;

  if (tid < 128) {
    int gr = mt * 128 + tid;
    tokS[tid] = (gr < rows) ? tok_list[e * NTOK + gr] : 0;
    wgtS[tid] = (gr < rows) ? wgt_list[e * NTOK + gr] : 0.0f;
  }
  __syncthreads();

  v8f acc[4][4];
  const v8f vzero = {0.f, 0.f, 0.f, 0.f, 0.f, 0.f, 0.f, 0.f};
#pragma unroll
  for (int i = 0; i < 4; ++i)
#pragma unroll
    for (int j = 0; j < 4; ++j) acc[i][j] = vzero;

  const int aj = tid & 7;
  const int ar = tid >> 3;
  const int bq = tid >> 6;
  const int bc = (tid & 63) * 4;

  float4 bReg[2][4];

  // async copy of one 128x32 bf16 A tile into LDS buffer `buf`.
  // Row clamped so every wave issues exactly 4 async ops per tile
  // (keeps s_wait_asynccnt thresholds exact, no divergence).
  auto issueA = [&](int buf, int k0) {
#pragma unroll
    for (int rr = 0; rr < 4; ++rr) {
      int r   = ar + rr * 32;
      int mg  = mt * 128 + r;
      int mgc = (mg < rows) ? mg : (rows - 1);
      unsigned int lds =
          (unsigned int)(unsigned long long)(void*)&As2[buf][r * 40 + 4 * aj];
      unsigned long long ga =
          (unsigned long long)(const void*)&hbuf[(size_t)(obase + mgc) * HDIM + k0 + 4 * aj];
      asm volatile("global_load_async_to_lds_b64 %0, %1, off"
                   :: "v"(lds), "v"(ga) : "memory");
    }
  };
  auto loadB = [&](int k0) {
#pragma unroll
    for (int it = 0; it < 2; ++it) {
      int kq = bq + it * 4;
#pragma unroll
      for (int kk = 0; kk < 4; ++kk)
        bReg[it][kk] =
            *(const float4*)&w2[((size_t)e * HDIM + (k0 + kq * 4 + kk)) * DDIM + nbase + bc];
    }
  };
  auto storeB = [&]() {
#pragma unroll
    for (int it = 0; it < 2; ++it) {
      int kq = bq + it * 4;
      const float* f0 = (const float*)&bReg[it][0];
      const float* f1 = (const float*)&bReg[it][1];
      const float* f2 = (const float*)&bReg[it][2];
      const float* f3 = (const float*)&bReg[it][3];
#pragma unroll
      for (int j = 0; j < 4; ++j) {
        uint2 p;
        p.x = pkbf(f0[j], f1[j]);
        p.y = pkbf(f2[j], f3[j]);
        *(uint2*)&Bs[(bc + j) * 40 + kq * 4] = p;
      }
    }
  };

  issueA(0, 0);
  loadB(0);
  for (int k0 = 0; k0 < HDIM; k0 += 32) {
    const int cur = (k0 >> 5) & 1;
    __syncthreads();          // prev compute done: other A buffer + Bs reusable
    storeB();
    const bool more = (k0 + 32) < HDIM;
    if (more) { issueA(cur ^ 1, k0 + 32); loadB(k0 + 32); }
    if (more) asm volatile("s_wait_asynccnt 0x4" ::: "memory");  // cur tile landed
    else      asm volatile("s_wait_asynccnt 0x0" ::: "memory");
    __syncthreads();          // all waves' async data + Bs visible

    FragU af[4], bfr[4];
    const int ml  = lane & 15;
    const int kba = (lane < 16) ? 0 : 8;
    const int kbb = (lane < 16) ? 0 : 8;
    const unsigned short* Ab = &As2[cur][0];
#pragma unroll
    for (int t = 0; t < 4; ++t) {
      int m = waveM + t * 16 + ml;
      af[t].u[0] = *(const uint4*)&Ab[m * 40 + kba];
      af[t].u[1] = *(const uint4*)&Ab[m * 40 + kba + 16];
      int n = waveN + t * 16 + ml;
      bfr[t].u[0] = *(const uint4*)&Bs[n * 40 + kbb];
      bfr[t].u[1] = *(const uint4*)&Bs[n * 40 + kbb + 16];
    }
#pragma unroll
    for (int ti = 0; ti < 4; ++ti)
#pragma unroll
      for (int tj = 0; tj < 4; ++tj)
        acc[ti][tj] = __builtin_amdgcn_wmma_f32_16x16x32_bf16(
            false, af[ti].v, false, bfr[tj].v, (short)0, acc[ti][tj], false, false);
  }

  const int ncol = lane & 15;
  const int mofs = (lane < 16) ? 0 : 8;
#pragma unroll
  for (int tj = 0; tj < 4; ++tj) {
    int ng     = nbase + waveN + tj * 16 + ncol;
    float bias = b2[(size_t)e * DDIM + ng];
#pragma unroll
    for (int ti = 0; ti < 4; ++ti) {
#pragma unroll
      for (int r = 0; r < 8; ++r) {
        int mloc = waveM + ti * 16 + r + mofs;
        int mg   = mt * 128 + mloc;
        if (mg < rows) {
          float v = wgtS[mloc] * (acc[ti][tj][r] + bias);
          atomicAdd(&out[(size_t)tokS[mloc] * DDIM + ng], v);
        }
      }
    }
  }
}

// ---------------------------------------------------------------- launch
extern "C" void kernel_launch(void* const* d_in, const int* in_sizes, int n_in,
                              void* d_out, int out_size, void* d_ws, size_t ws_size,
                              hipStream_t stream) {
  (void)in_sizes; (void)n_in; (void)out_size; (void)ws_size;
  const float* x  = (const float*)d_in[0];
  const float* gw = (const float*)d_in[1];
  const float* gb = (const float*)d_in[2];
  const float* w1 = (const float*)d_in[3];
  const float* b1 = (const float*)d_in[4];
  const float* w2 = (const float*)d_in[5];
  const float* b2 = (const float*)d_in[6];
  float* out = (float*)d_out;

  int*   counts   = (int*)d_ws;
  int*   offsets  = counts + 8;
  int*   tok_list = offsets + 8;
  float* wgt_list = (float*)(tok_list + NEXP * NTOK);
  unsigned short* hbuf = (unsigned short*)(wgt_list + NEXP * NTOK);  // 16384 x 4096 bf16

  zero_out_kernel<<<4096, 256, 0, stream>>>(out, counts, NTOK * DDIM);
  gate_topk_kernel<<<NTOK / 8, 256, 0, stream>>>(x, gw, gb, counts, tok_list, wgt_list);
  prefix_kernel<<<1, 32, 0, stream>>>(counts, offsets);

  dim3 g1(HDIM / 256, NTOK / 128, NEXP);
  gemm1_gelu_kernel<<<g1, 256, 0, stream>>>(x, w1, b1, counts, offsets, tok_list, hbuf);

  dim3 g2(DDIM / 256, NTOK / 128, NEXP);
  gemm2_scatter_kernel<<<g2, 256, 0, stream>>>(hbuf, w2, b2, counts, offsets,
                                               tok_list, wgt_list, out);
}
